// StackedRNNLayer_45131516346883
// MI455X (gfx1250) — matrix operand order repaired and checked by reading, not compiled, not run on hardware
//
#include <hip/hip_runtime.h>

#define NWG   32
#define NTHR  256

typedef __bf16 bf16_t;
typedef bf16_t v16bf __attribute__((ext_vector_type(16)));
typedef float  v8f   __attribute__((ext_vector_type(8)));

constexpr int Tt = 256, Bb = 64, Dd = 1024, Ll = 4, Kg = 2048;
constexpr long long WG_ELEMS = (long long)Ll * 3 * Kg * Dd;   // bf16 elements (gates)
constexpr long long WP_ELEMS = (long long)Ll * Dd * Dd;       // bf16 elements (proj)
constexpr size_t    WG_BYTES = (size_t)WG_ELEMS * 2;
constexpr size_t    WP_BYTES = (size_t)WP_ELEMS * 2;

constexpr int KL_B_STRIDE   = 64 * 32 * 16;   // bf16 elems between consecutive k-tiles of B
constexpr int KL_LDS_STRIDE = 4 * 32 * 16;    // bf16 elems between consecutive k-tiles of A in LDS

__device__ __forceinline__ unsigned short f2bf(float f) {
  unsigned x = __float_as_uint(f);
  x += 0x7FFFu + ((x >> 16) & 1u);              // round-to-nearest-even
  return (unsigned short)(x >> 16);
}

__device__ __forceinline__ v8f wmma_bf16(v16bf a, v16bf b, v8f c) {
  return __builtin_amdgcn_wmma_f32_16x16x32_bf16(
      false, a, false, b, (short)0, c, false, false);
}

__device__ __forceinline__ float sigmoidf_(float x) {
  return 1.0f / (1.0f + __expf(-x));
}

__device__ void grid_sync(unsigned* sync) {
  __syncthreads();
  if (threadIdx.x == 0) {
    __threadfence();
    unsigned gen = __atomic_load_n(&sync[1], __ATOMIC_ACQUIRE);
    unsigned arr = __atomic_fetch_add(&sync[0], 1u, __ATOMIC_ACQ_REL);
    if (arr == (unsigned)(NWG - 1)) {
      __atomic_store_n(&sync[0], 0u, __ATOMIC_RELAXED);
      __threadfence();
      __atomic_fetch_add(&sync[1], 1u, __ATOMIC_RELEASE);
    } else {
      while (__atomic_load_n(&sync[1], __ATOMIC_ACQUIRE) == gen)
        __builtin_amdgcn_s_sleep(1);
    }
    __threadfence();
  }
  __syncthreads();
}

// -------- weight conversion fp32 -> bf16, pre-swizzled to WMMA B-fragment layout --------
// B frag (16-bit, 32x16): lane = (khalf<<4)|ncol ; element e -> K = kt*32 + khalf*16 + e,
// N = nt*16 + ncol.  Storage: [tile(kt,nt)][lane][16 bf16] -> 32B contiguous per lane.
__global__ void prep_kernel(const float* __restrict__ Wr, const float* __restrict__ Wu,
                            const float* __restrict__ Wn, const float* __restrict__ Wp,
                            unsigned short* __restrict__ wg_s, unsigned short* __restrict__ wp_s,
                            float* __restrict__ m_s, float* __restrict__ c_s,
                            unsigned* __restrict__ sync)
{
  long long idx = (long long)blockIdx.x * blockDim.x + threadIdx.x;
  long long stride = (long long)gridDim.x * blockDim.x;
  if (idx == 0) { sync[0] = 0u; sync[1] = 0u; }
  for (long long i = idx; i < (long long)Bb * Dd; i += stride) { m_s[i] = 0.f; c_s[i] = 0.f; }

  for (long long d = idx; d < WG_ELEMS; d += stride) {
    int e    = (int)(d & 15);
    long long q = d >> 4;
    int lane = (int)(q & 31); q >>= 5;
    int nt   = (int)(q & 63); q >>= 6;
    int kt   = (int)(q & 63); q >>= 6;
    int g    = (int)(q % 3);
    int l    = (int)(q / 3);
    int k = kt * 32 + ((lane >> 4) << 4) + e;
    int n = nt * 16 + (lane & 15);
    const float* Ws = (g == 0) ? Wr : ((g == 1) ? Wu : Wn);
    wg_s[d] = f2bf(Ws[((long long)l * Kg + k) * Dd + n]);
  }
  for (long long d = idx; d < WP_ELEMS; d += stride) {
    int e    = (int)(d & 15);
    long long q = d >> 4;
    int lane = (int)(q & 31); q >>= 5;
    int nt   = (int)(q & 63); q >>= 6;
    int kt   = (int)(q & 31); q >>= 5;
    int l    = (int)q;
    int k = kt * 32 + ((lane >> 4) << 4) + e;
    int n = nt * 16 + (lane & 15);
    wp_s[d] = f2bf(Wp[((long long)l * Dd + k) * Dd + n]);
  }
}

// -------- stage a 512-wide K chunk of the A matrix (64 rows) into LDS A-frag layout --------
// A frag (16-bit, 16x32): lane = (half<<4)|mrow, VGPR v holds K = half*16 + v*2 (+0/1)
// mode 0: [x | m], mode 1: [x | r*m], mode 2: c_new
__device__ __forceinline__ void stage_chunk(unsigned* ldsU, int ktc, int mode,
                                            const float* __restrict__ xrow,
                                            const float* m_s, const float* r_s, const float* c_s)
{
  for (int p = threadIdx.x; p < 16384; p += NTHR) {
    int kp = p & 255;          // k-pair within chunk
    int m  = p >> 8;           // batch row 0..63
    int kc = kp << 1;          // even k within chunk
    int kg = ktc * 512 + kc;   // global k
    float v0, v1;
    if (mode == 2) {
      v0 = c_s[m * Dd + kg];  v1 = c_s[m * Dd + kg + 1];
    } else if (kg < Dd) {
      v0 = xrow[m * Dd + kg]; v1 = xrow[m * Dd + kg + 1];
    } else {
      int j = kg - Dd;
      float m0 = m_s[m * Dd + j], m1 = m_s[m * Dd + j + 1];
      if (mode == 0) { v0 = m0; v1 = m1; }
      else           { v0 = r_s[m * Dd + j] * m0; v1 = r_s[m * Dd + j + 1] * m1; }
    }
    int ktl  = kc >> 5;
    int kk   = kc & 31;
    int vi   = (kk & 15) >> 1;  // VGPR index: (k within half) / 2
    int half = (kk >> 4) & 1;
    int lanei = (half << 4) | (m & 15);
    int mt    = m >> 4;
    unsigned pk = (unsigned)f2bf(v0) | ((unsigned)f2bf(v1) << 16);
    ldsU[((ktl * 4 + mt) * 32 + lanei) * 8 + vi] = pk;
  }
}

__global__ void __launch_bounds__(NTHR)
rnn_kernel(const float* __restrict__ x_in, const float* __restrict__ pad,
           const float* __restrict__ br, const float* __restrict__ bu,
           const float* __restrict__ bn,
           const unsigned short* __restrict__ wg_s, const unsigned short* __restrict__ wp_s,
           float* m_s, float* c_s, float* r_s, float* u_s,
           float* xs_out, float* st_m, float* st_c, unsigned* sync)
{
  __shared__ __align__(64) unsigned short ldsA[16 * 4 * 32 * 16];  // 64 KB chunk of A frags
  unsigned* ldsU = (unsigned*)ldsA;
  const int tid  = threadIdx.x;
  const int wg   = blockIdx.x;
  const int wv   = tid >> 5;
  const int lane = tid & 31;
  const int nl   = lane & 15;
  const int hh   = lane >> 4;

  for (int l = 0; l < Ll; ++l) {
    const float* xsrc = (l == 0) ? x_in : xs_out;
    const unsigned short* Wr_s = wg_s + (long long)(l * 3 + 0) * (Kg * Dd);
    const unsigned short* Wu_s = wg_s + (long long)(l * 3 + 1) * (Kg * Dd);
    const unsigned short* Wn_s = wg_s + (long long)(l * 3 + 2) * (Kg * Dd);
    const unsigned short* Wpl  = wp_s + (long long)l * (Dd * Dd);

    for (int t = 0; t < Tt; ++t) {
      const float* xrow = xsrc + (long long)t * Bb * Dd;
      const float* prow = pad + t * Bb;

      // ================= Phase 1: r,u = sigmoid([x|m] @ W + b) =================
      {
        const int gate  = wg >> 4;                  // WG 0-15: r, WG 16-31: u
        const int ncol0 = (wg & 15) * 64;
        const unsigned short* Wsel = gate ? Wu_s : Wr_s;
        const float* bias = gate ? bu : br;
        float* dst = gate ? u_s : r_s;
        const int mt   = wv & 3;
        const int ntl0 = (wv >> 2) << 1;
        const int ntg0 = ((wg & 15) << 2) + ntl0;
        const unsigned short* ldsWv = ldsA + (mt * 32 + lane) * 16;  // + kl*KL_LDS_STRIDE
        v8f acc0 = {0.f,0.f,0.f,0.f,0.f,0.f,0.f,0.f};
        v8f acc1 = {0.f,0.f,0.f,0.f,0.f,0.f,0.f,0.f};
        for (int ktc = 0; ktc < 4; ++ktc) {
          __syncthreads();
          stage_chunk(ldsU, ktc, 0, xrow, m_s, r_s, c_s);
          __syncthreads();
          // chunk base for B: all 16 k-tiles reachable by constant immediate offsets
          const unsigned short* bpB =
              Wsel + (((long long)(ktc * 16 * 64 + ntg0)) * 32 + lane) * 16;
          if (ktc + 1 < 4)
            __builtin_prefetch(bpB + 16LL * KL_B_STRIDE, 0, 1);
          // 1-deep software pipeline: prefetch (kl+1) frags before WMMAs of kl
          v16bf a  = *(const v16bf*)(ldsWv);
          v16bf b0 = *(const v16bf*)(bpB);
          v16bf b1 = *(const v16bf*)(bpB + 512);
          #pragma unroll
          for (int kl = 0; kl < 16; ++kl) {
            v16bf an, bn0, bn1;
            if (kl < 15) {
              an  = *(const v16bf*)(ldsWv + (kl + 1) * KL_LDS_STRIDE);
              bn0 = *(const v16bf*)(bpB + (kl + 1) * KL_B_STRIDE);
              bn1 = *(const v16bf*)(bpB + (kl + 1) * KL_B_STRIDE + 512);
            }
            acc0 = wmma_bf16(a, b0, acc0);
            acc1 = wmma_bf16(a, b1, acc1);
            if (kl < 15) { a = an; b0 = bn0; b1 = bn1; }
          }
        }
        #pragma unroll
        for (int rr = 0; rr < 8; ++rr) {
          int row = mt * 16 + hh * 8 + rr;           // C/D layout: VGPR rr -> M=rr (+8 for hi lanes)
          int c0  = ncol0 + ntl0 * 16 + nl;
          int c1  = c0 + 16;
          dst[row * Dd + c0] = sigmoidf_(acc0[rr] + bias[l * Dd + c0]);
          dst[row * Dd + c1] = sigmoidf_(acc1[rr] + bias[l * Dd + c1]);
        }
      }
      grid_sync(sync);

      // ============ Phase 2: n = tanh([x|r*m] @ Wn + bn); c update ============
      {
        const int ncol0 = wg * 32;
        const int mt  = wv & 3;
        const int ntl = wv >> 2;                    // 0..1
        const int ntg = wg * 2 + ntl;
        const unsigned short* ldsWv = ldsA + (mt * 32 + lane) * 16;
        v8f acc = {0.f,0.f,0.f,0.f,0.f,0.f,0.f,0.f};
        for (int ktc = 0; ktc < 4; ++ktc) {
          __syncthreads();
          stage_chunk(ldsU, ktc, 1, xrow, m_s, r_s, c_s);
          __syncthreads();
          const unsigned short* bpB =
              Wn_s + (((long long)(ktc * 16 * 64 + ntg)) * 32 + lane) * 16;
          if (ktc + 1 < 4)
            __builtin_prefetch(bpB + 16LL * KL_B_STRIDE, 0, 1);
          v16bf a = *(const v16bf*)(ldsWv);
          v16bf b = *(const v16bf*)(bpB);
          #pragma unroll
          for (int kl = 0; kl < 16; ++kl) {
            v16bf an, bn1;
            if (kl < 15) {
              an  = *(const v16bf*)(ldsWv + (kl + 1) * KL_LDS_STRIDE);
              bn1 = *(const v16bf*)(bpB + (kl + 1) * KL_B_STRIDE);
            }
            acc = wmma_bf16(a, b, acc);
            if (kl < 15) { a = an; b = bn1; }
          }
        }
        const int col = ncol0 + ntl * 16 + nl;
        const float bnv = bn[l * Dd + col];
        #pragma unroll
        for (int rr = 0; rr < 8; ++rr) {
          int row = mt * 16 + hh * 8 + rr;
          float nv = tanhf(acc[rr] + bnv);
          float uu = u_s[row * Dd + col];
          float cc = c_s[row * Dd + col];
          float cn = (1.f - uu) * nv + uu * cc;
          float pv = prow[row];
          cn = pv * cc + (1.f - pv) * cn;
          c_s[row * Dd + col] = cn;
          if (t == Tt - 1) st_c[(l * Bb + row) * Dd + col] = cn;
        }
      }
      grid_sync(sync);

      // ================= Phase 3: m_new = c_new @ Wp; residual ================
      {
        const int ncol0 = wg * 32;
        const int mt  = wv & 3;
        const int ntl = wv >> 2;
        const int ntg = wg * 2 + ntl;
        const unsigned short* ldsWv = ldsA + (mt * 32 + lane) * 16;
        v8f acc = {0.f,0.f,0.f,0.f,0.f,0.f,0.f,0.f};
        for (int ktc = 0; ktc < 2; ++ktc) {
          __syncthreads();
          stage_chunk(ldsU, ktc, 2, xrow, m_s, r_s, c_s);
          __syncthreads();
          const unsigned short* bpB =
              Wpl + (((long long)(ktc * 16 * 64 + ntg)) * 32 + lane) * 16;
          if (ktc + 1 < 2)
            __builtin_prefetch(bpB + 16LL * KL_B_STRIDE, 0, 1);
          v16bf a = *(const v16bf*)(ldsWv);
          v16bf b = *(const v16bf*)(bpB);
          #pragma unroll
          for (int kl = 0; kl < 16; ++kl) {
            v16bf an, bn1;
            if (kl < 15) {
              an  = *(const v16bf*)(ldsWv + (kl + 1) * KL_LDS_STRIDE);
              bn1 = *(const v16bf*)(bpB + (kl + 1) * KL_B_STRIDE);
            }
            acc = wmma_bf16(a, b, acc);
            if (kl < 15) { a = an; b = bn1; }
          }
        }
        const int col = ncol0 + ntl * 16 + nl;
        #pragma unroll
        for (int rr = 0; rr < 8; ++rr) {
          int row = mt * 16 + hh * 8 + rr;
          float mr = acc[rr];
          float mo = m_s[row * Dd + col];
          float pv = prow[row];
          float mn = pv * mo + (1.f - pv) * mr;
          m_s[row * Dd + col] = mn;
          long long xi = ((long long)t * Bb + row) * Dd + col;
          float xv = (l == 0) ? mn : (xs_out[xi] + mn);
          xs_out[xi] = xv;
          if (t == Tt - 1) st_m[(l * Bb + row) * Dd + col] = mn;
        }
      }
      grid_sync(sync);
    }
  }
}

extern "C" void kernel_launch(void* const* d_in, const int* in_sizes, int n_in,
                              void* d_out, int out_size, void* d_ws, size_t ws_size,
                              hipStream_t stream)
{
  (void)in_sizes; (void)n_in; (void)out_size; (void)ws_size;
  const float* x   = (const float*)d_in[0];
  const float* pad = (const float*)d_in[1];
  const float* Wr  = (const float*)d_in[2];
  const float* Wu  = (const float*)d_in[3];
  const float* Wn  = (const float*)d_in[4];
  const float* br  = (const float*)d_in[5];
  const float* bu  = (const float*)d_in[6];
  const float* bn  = (const float*)d_in[7];
  const float* Wp  = (const float*)d_in[8];

  char* ws = (char*)d_ws;
  unsigned*        sync = (unsigned*)ws;
  unsigned short*  wg_s = (unsigned short*)(ws + 256);
  unsigned short*  wp_s = (unsigned short*)(ws + 256 + WG_BYTES);
  float* m_s = (float*)(ws + 256 + WG_BYTES + WP_BYTES);
  float* c_s = m_s + (size_t)Bb * Dd;
  float* r_s = c_s + (size_t)Bb * Dd;
  float* u_s = r_s + (size_t)Bb * Dd;

  float* xs_out = (float*)d_out;
  float* st_m   = xs_out + (long long)Tt * Bb * Dd;
  float* st_c   = st_m + (long long)Ll * Bb * Dd;

  prep_kernel<<<2048, NTHR, 0, stream>>>(Wr, Wu, Wn, Wp, wg_s, wp_s, m_s, c_s, sync);
  rnn_kernel<<<NWG, NTHR, 0, stream>>>(x, pad, br, bu, bn, wg_s, wp_s,
                                       m_s, c_s, r_s, u_s, xs_out, st_m, st_c, sync);
}
